// CondNet_15135464751285
// MI455X (gfx1250) — matrix operand order repaired and checked by reading, not compiled, not run on hardware
//
#include <hip/hip_runtime.h>
#include <hip/hip_bf16.h>
#include <stdint.h>
#include <stddef.h>

// Problem constants (match reference)
#define Nn    10000
#define Hh    64
#define Pp    12
#define NBl   3
#define INch  3
#define Ee    160000
#define Bb    2
#define BPc   (Bb*Pp)        // 24
#define ROWS  (BPc*Nn)       // 240000
#define SLICE (Nn*Hh)        // 640000 elems per (b,p) LN slice
#define SBLK  625            // stats partial blocks per slice (625*1024 == 640000)
#define EPSf  1e-5f

typedef __attribute__((ext_vector_type(16))) _Float16     v16h;
typedef __attribute__((ext_vector_type(8)))  float        v8f;
typedef __attribute__((ext_vector_type(4)))  unsigned int u32x4;
typedef __attribute__((ext_vector_type(8)))  int          i32x8;
typedef __attribute__((ext_vector_type(4)))  int          i32x4;

// ---------------------------------------------------------------------------
// TDM: stage a 64x64 f16 tile from global memory into LDS (D# per ISA ch.8).
// amdgpu-toolchain (clang-23) 6-arg form: (g0, g1, g2, g3, g?, cpol).
// ---------------------------------------------------------------------------
__device__ __forceinline__ void tdm_load_w64(const _Float16* g, _Float16* lds) {
  unsigned long long ga = (unsigned long long)(uintptr_t)g;
  unsigned int ldsoff  = (unsigned int)(uintptr_t)lds;  // LDS aperture: offset lives in addr[31:0]
  u32x4 g0;
  g0[0] = 1u;                                               // count=1, user descriptor
  g0[1] = ldsoff;                                           // lds_addr (bytes)
  g0[2] = (unsigned int)ga;                                 // global_addr[31:0]
  g0[3] = ((unsigned int)(ga >> 32) & 0x01FFFFFFu) | (2u << 30); // addr[56:32] | type=2
  i32x8 g1;
  g1[0] = 1 << 16;        // data_size = 2 bytes (f16)
  g1[1] = 64 << 16;       // tensor_dim0 = 64
  g1[2] = 64 << 16;       // tensor_dim0 hi=0 | tensor_dim1 = 64
  g1[3] = 64 << 16;       // tensor_dim1 hi=0 | tile_dim0 = 64
  g1[4] = 64;             // tile_dim1 = 64, tile_dim2 = 0
  g1[5] = 64;             // tensor_dim0_stride = 64 elements
  g1[6] = 4096 << 16;     // tensor_dim1_stride = 4096 (unused for 2D tile)
  g1[7] = 0;
  i32x4 z4 = {};
  i32x8 z8 = {};
  __builtin_amdgcn_tensor_load_to_lds(g0, g1, z4, z4, z8, 0);
}

// ---------------------------------------------------------------------------
// WMMA fragment assembly from LDS (contiguous k = 16*(lane/16) + s in A and B, wave32)
// ---------------------------------------------------------------------------
// A fragment: 16x32 f16. lanes 0-15 -> K {k0..k0+15}; lanes 16-31
// hold K {k0+16..k0+31} of the same rows.
__device__ __forceinline__ v16h frag_a_lds(const _Float16* A, int lda, int row0, int k0) {
  const int lane = threadIdx.x & 31;
  const int m    = lane & 15;
  const int kb   = (lane >> 4) * 16;
  const _Float16* p = A + (row0 + m) * lda + k0;
  v16h a;
#pragma unroll
  for (int h = 0; h < 8; ++h) a[h] = p[kb + h];
#pragma unroll
  for (int h = 0; h < 8; ++h) a[8 + h] = p[8 + kb + h];
  return a;
}

// B fragment: 32x16 f16, B[k][n] = W[n][k] (weight stored row-major [out][in]).
// VGPR v holds rows K=2v,2v+1 striped across 16 lanes; lanes 16-31 do K+16.
__device__ __forceinline__ v16h frag_b_ldsT(const _Float16* W, int k0, int n0) {
  const int lane = threadIdx.x & 31;
  const int n    = n0 + (lane & 15);
  const int kb   = k0 + (lane >> 4) * 16;
  v16h b;
#pragma unroll
  for (int h = 0; h < 16; ++h) b[h] = W[n * 64 + kb + h];
  return b;
}

// One wave computes a 16(M) x 64(N) strip, K=64 in two 32-deep WMMA steps.
__device__ __forceinline__ void wave_gemm_16x64(const _Float16* A, int lda, int arow0,
                                                const _Float16* W, v8f c[4]) {
#pragma unroll
  for (int k0 = 0; k0 < 64; k0 += 32) {
    v16h a = frag_a_lds(A, lda, arow0, k0);
#pragma unroll
    for (int nt = 0; nt < 4; ++nt) {
      v16h b = frag_b_ldsT(W, k0, nt * 16);
      c[nt] = __builtin_amdgcn_wmma_f32_16x16x32_f16(
          /*neg_a=*/false, a, /*neg_b=*/false, b,
          /*c_mod=*/(short)0, c[nt], /*reuse_a=*/false, /*reuse_b=*/false);
    }
  }
}

// softmax over the 4 attention logits, recomputed inline (4 loads, 4 exp)
__device__ __forceinline__ float attn_w(const float* attn, int k) {
  float a0 = attn[0], a1 = attn[1], a2 = attn[2], a3 = attn[3];
  float m  = fmaxf(fmaxf(a0, a1), fmaxf(a2, a3));
  float e0 = __expf(a0 - m), e1 = __expf(a1 - m), e2 = __expf(a2 - m), e3 = __expf(a3 - m);
  float s  = e0 + e1 + e2 + e3;
  float ek = (k == 0) ? e0 : (k == 1) ? e1 : (k == 2) ? e2 : e3;
  return ek / s;
}

// ---------------------------------------------------------------------------
// Small utility kernels
// ---------------------------------------------------------------------------
__global__ void k_zero(float* __restrict__ p, int n) {
  int i = blockIdx.x * 256 + threadIdx.x;
  if (i < n) p[i] = 0.f;
}

__global__ void k_cvt_w(const float* __restrict__ iw2, const float* __restrict__ pw1,
                        const float* __restrict__ pw2, const float* __restrict__ ow1,
                        const float* __restrict__ ow2, _Float16* __restrict__ wf) {
  int i = blockIdx.x * 256 + threadIdx.x;
  if (i >= 5 * 4096) return;
  const float* srcs[5] = {iw2, pw1, pw2, ow1, ow2};
  wf[i] = (_Float16)srcs[i >> 12][i & 4095];
}

// ---------------------------------------------------------------------------
// Graph preprocessing: degree -> norm -> CSR by destination (col)
// ---------------------------------------------------------------------------
__global__ void k_deg(const long long* __restrict__ ei, float* __restrict__ deg) {
  int e = blockIdx.x * 256 + threadIdx.x;
  if (e < Ee) atomicAdd(&deg[(int)ei[Ee + e]], 1.f);
}

__global__ void k_norm(const long long* __restrict__ ei, const float* __restrict__ deg,
                       float* __restrict__ nrm) {
  int e = blockIdx.x * 256 + threadIdx.x;
  if (e >= Ee) return;
  float dr = deg[(int)ei[e]], dc = deg[(int)ei[Ee + e]];
  nrm[e] = (dr > 0.f && dc > 0.f) ? rsqrtf(dr) * rsqrtf(dc) : 0.f;
}

__global__ __launch_bounds__(1024) void k_scan(const float* __restrict__ deg,
                                               int* __restrict__ rptr) {
  __shared__ int part[1024];
  const int tid = threadIdx.x;
  const int CH  = (Nn + 1023) / 1024;  // 10
  const int base = tid * CH;
  int s = 0;
  for (int i = 0; i < CH; ++i) {
    int idx = base + i;
    if (idx < Nn) s += (int)deg[idx];
  }
  part[tid] = s;
  __syncthreads();
  for (int off = 1; off < 1024; off <<= 1) {
    int v = (tid >= off) ? part[tid - off] : 0;
    __syncthreads();
    part[tid] += v;
    __syncthreads();
  }
  int run = (tid == 0) ? 0 : part[tid - 1];
  for (int i = 0; i < CH; ++i) {
    int idx = base + i;
    if (idx < Nn) { rptr[idx] = run; run += (int)deg[idx]; }
  }
  if (tid == 1023) rptr[Nn] = run;
}

__global__ void k_csr_init(const int* __restrict__ rptr, int* __restrict__ cur) {
  int n = blockIdx.x * 256 + threadIdx.x;
  if (n < Nn) cur[n] = rptr[n];
}

__global__ void k_fill(const long long* __restrict__ ei, const float* __restrict__ nrm,
                       int* __restrict__ cur, int* __restrict__ csrc, float* __restrict__ cw) {
  int e = blockIdx.x * 256 + threadIdx.x;
  if (e >= Ee) return;
  int c = (int)ei[Ee + e];
  int p = atomicAdd(&cur[c], 1);
  csrc[p] = (int)ei[e];
  cw[p]   = nrm[e];
}

// ---------------------------------------------------------------------------
// pos projection: posP = relu(pos@pw1^T + pb1) @ pw2^T + pb2   (WMMA, TDM)
// ---------------------------------------------------------------------------
__global__ __launch_bounds__(256)
void k_pos_proj(const float* __restrict__ pos,
                const _Float16* __restrict__ w1, const float* __restrict__ b1,
                const _Float16* __restrict__ w2, const float* __restrict__ b2,
                float* __restrict__ posP) {
  __shared__ _Float16 sW1[64 * 64];
  __shared__ _Float16 sW2[64 * 64];
  __shared__ _Float16 sA[128 * 72];
  __shared__ _Float16 sH[128 * 72];
  const int tid = threadIdx.x;
  if (tid == 0) { tdm_load_w64(w1, sW1); tdm_load_w64(w2, sW2); }
  const int row0 = blockIdx.x * 128;
  for (int idx = tid; idx < 128 * 64; idx += 256) {
    int r = idx >> 6, cc = idx & 63, rr = row0 + r;
    sA[r * 72 + cc] = (rr < Nn) ? (_Float16)pos[(size_t)rr * 64 + cc] : (_Float16)0.f;
  }
  __builtin_amdgcn_s_wait_tensorcnt(0);
  __syncthreads();
  const int wave = tid >> 5, lane = tid & 31;
  const int arow0 = wave * 16, half = lane >> 4, nlo = lane & 15;
  v8f c[4] = {};
  wave_gemm_16x64(sA, 72, arow0, sW1, c);
#pragma unroll
  for (int nt = 0; nt < 4; ++nt) {
#pragma unroll
    for (int r = 0; r < 8; ++r) {
      int rl = arow0 + half * 8 + r, col = nt * 16 + nlo;
      sH[rl * 72 + col] = (_Float16)fmaxf(c[nt][r] + b1[col], 0.f);
    }
  }
  __syncthreads();
  v8f d[4] = {};
  wave_gemm_16x64(sH, 72, arow0, sW2, d);
#pragma unroll
  for (int nt = 0; nt < 4; ++nt) {
#pragma unroll
    for (int r = 0; r < 8; ++r) {
      int rl = arow0 + half * 8 + r, col = nt * 16 + nlo, rr = row0 + rl;
      if (rr < Nn) posP[(size_t)rr * 64 + col] = d[nt][r] + b2[col];
    }
  }
}

// ---------------------------------------------------------------------------
// input projection: X = relu(x@iw1^T+ib1)@iw2^T + ib2 + posP ; ACC = a0 * X
// ---------------------------------------------------------------------------
__global__ __launch_bounds__(256)
void k_proj_in(const float* __restrict__ x,
               const float* __restrict__ iw1, const float* __restrict__ ib1,
               const _Float16* __restrict__ w2, const float* __restrict__ ib2,
               const float* __restrict__ posP, const float* __restrict__ attn,
               float* __restrict__ X, float* __restrict__ ACC) {
  __shared__ _Float16 sW2[64 * 64];
  __shared__ _Float16 sH[128 * 72];
  __shared__ float    sX[128 * 4];
  const int tid = threadIdx.x;
  if (tid == 0) tdm_load_w64(w2, sW2);
  const int row0 = blockIdx.x * 128;  // 240000 rows, 1875 blocks, exact
  for (int idx = tid; idx < 128 * INch; idx += 256) {
    int r = idx / INch, i = idx % INch;
    sX[r * 4 + i] = x[(size_t)(row0 + r) * INch + i];
  }
  __syncthreads();
  // layer1: K=3, scalar
  for (int idx = tid; idx < 128 * 64; idx += 256) {
    int r = idx >> 6, o = idx & 63;
    float s = ib1[o];
#pragma unroll
    for (int i = 0; i < INch; ++i) s += sX[r * 4 + i] * iw1[o * INch + i];
    sH[r * 72 + o] = (_Float16)fmaxf(s, 0.f);
  }
  __builtin_amdgcn_s_wait_tensorcnt(0);
  __syncthreads();
  const int wave = tid >> 5, lane = tid & 31;
  const int arow0 = wave * 16, half = lane >> 4, nlo = lane & 15;
  v8f c[4] = {};
  wave_gemm_16x64(sH, 72, arow0, sW2, c);
  const float a0 = attn_w(attn, 0);
#pragma unroll
  for (int nt = 0; nt < 4; ++nt) {
#pragma unroll
    for (int r = 0; r < 8; ++r) {
      int rl = arow0 + half * 8 + r, col = nt * 16 + nlo;
      int row = row0 + rl, n = row % Nn;
      float v = c[nt][r] + ib2[col] + posP[(size_t)n * 64 + col];
      size_t o = (size_t)row * 64 + col;
      X[o]   = v;
      ACC[o] = a0 * v;
    }
  }
}

// ---------------------------------------------------------------------------
// LayerNorm stats over each (b,p) slice: deterministic 2-pass reduction
// ---------------------------------------------------------------------------
__global__ __launch_bounds__(256)
void k_stats(const float* __restrict__ Xv, float* __restrict__ part) {
  const int bp = blockIdx.y;
  const float* base = Xv + (size_t)bp * SLICE;
  const int start = blockIdx.x * 1024 + threadIdx.x;
  float s = 0.f, ss = 0.f;
#pragma unroll
  for (int i = 0; i < 4; ++i) {
    float v = base[start + i * 256];
    s += v; ss += v * v;
  }
#pragma unroll
  for (int o = 16; o > 0; o >>= 1) { s += __shfl_down(s, o, 32); ss += __shfl_down(ss, o, 32); }
  __shared__ float red[16];
  int wv = threadIdx.x >> 5, ln = threadIdx.x & 31;
  if (ln == 0) { red[wv] = s; red[8 + wv] = ss; }
  __syncthreads();
  if (threadIdx.x == 0) {
    float S = 0.f, SS = 0.f;
    for (int w = 0; w < 8; ++w) { S += red[w]; SS += red[8 + w]; }
    part[((size_t)bp * SBLK + blockIdx.x) * 2 + 0] = S;
    part[((size_t)bp * SBLK + blockIdx.x) * 2 + 1] = SS;
  }
}

__global__ __launch_bounds__(256)
void k_stats_final(const float* __restrict__ part, float* __restrict__ st) {
  const int bp = blockIdx.x;
  float s = 0.f, ss = 0.f;
  for (int j = threadIdx.x; j < SBLK; j += 256) {
    s  += part[((size_t)bp * SBLK + j) * 2 + 0];
    ss += part[((size_t)bp * SBLK + j) * 2 + 1];
  }
#pragma unroll
  for (int o = 16; o > 0; o >>= 1) { s += __shfl_down(s, o, 32); ss += __shfl_down(ss, o, 32); }
  __shared__ float red[16];
  int wv = threadIdx.x >> 5, ln = threadIdx.x & 31;
  if (ln == 0) { red[wv] = s; red[8 + wv] = ss; }
  __syncthreads();
  if (threadIdx.x == 0) {
    float S = 0.f, SS = 0.f;
    for (int w = 0; w < 8; ++w) { S += red[w]; SS += red[8 + w]; }
    st[bp * 2 + 0] = S;
    st[bp * 2 + 1] = SS;
  }
}

// ---------------------------------------------------------------------------
// LightGCN propagate with fused LayerNorm on the gathered source rows.
// CSR gather: block = (64 h-lanes) x (4 bp), grid = (N, 6). L2-resident reads.
// ---------------------------------------------------------------------------
__global__ __launch_bounds__(256)
void k_propagate(const float* __restrict__ X, const int* __restrict__ rptr,
                 const int* __restrict__ csrc, const float* __restrict__ cw,
                 const float* __restrict__ st,
                 const float* __restrict__ gnw, const float* __restrict__ gnb,
                 float* __restrict__ XB) {
  const int n  = blockIdx.x;
  const int h  = threadIdx.x & 63;
  const int bp = blockIdx.y * 4 + (threadIdx.x >> 6);
  const float inv = 1.f / (float)SLICE;
  const float mu  = st[bp * 2 + 0] * inv;
  const float var = st[bp * 2 + 1] * inv - mu * mu;
  const float rs  = rsqrtf(var + EPSf);
  const int s0 = rptr[n], s1 = rptr[n + 1];
  const float* Xbp = X + (size_t)bp * SLICE;
  float acc = 0.f;
  for (int j = s0; j < s1; ++j) {
    if (j + 1 < s1) __builtin_prefetch(&Xbp[(size_t)csrc[j + 1] * Hh + h], 0, 0);
    int   src = csrc[j];
    float w   = cw[j];
    float v   = Xbp[(size_t)src * Hh + h];
    acc += w * ((v - mu) * rs * gnw[src * Hh + h] + gnb[src * Hh + h]);
  }
  XB[((size_t)bp * Nn + n) * Hh + h] = acc;
}

// ---------------------------------------------------------------------------
// Fused: LayerNorm2 + 12x12 period conv + skip accumulation into ACC
// thread per (b,n,h); P=12 values mixed in registers.
// ---------------------------------------------------------------------------
__global__ __launch_bounds__(256)
void k_tconv(const float* __restrict__ XB, const float* __restrict__ st,
             const float* __restrict__ tnw, const float* __restrict__ tnb,
             const float* __restrict__ tcw, const float* __restrict__ tcb,
             const float* __restrict__ attn, int blk,
             float* __restrict__ Xn, float* __restrict__ ACC) {
  __shared__ float sW[Pp * Pp];
  __shared__ float sB[Pp];
  __shared__ float sMu[BPc], sRs[BPc];
  const int tid = threadIdx.x;
  if (tid < Pp * Pp) sW[tid] = tcw[tid];
  if (tid < Pp) sB[tid] = tcb[tid];
  if (tid < BPc) {
    const float inv = 1.f / (float)SLICE;
    float mu  = st[tid * 2 + 0] * inv;
    float var = st[tid * 2 + 1] * inv - mu * mu;
    sMu[tid] = mu;
    sRs[tid] = rsqrtf(var + EPSf);
  }
  __syncthreads();
  const int gid = blockIdx.x * 256 + tid;  // over B*N*H = 1,280,000 (exact grid)
  if (gid >= Bb * Nn * Hh) return;
  const int h = gid & 63;
  const int n = (gid >> 6) % Nn;
  const int b = gid / (Nn * Hh);
  const float w  = tnw[n * Hh + h];
  const float bb = tnb[n * Hh + h];
  float out[Pp];
#pragma unroll
  for (int q = 0; q < Pp; ++q) out[q] = sB[q];
  for (int p = 0; p < Pp; ++p) {
    const int bp = b * Pp + p;
    float v  = XB[((size_t)bp * Nn + n) * Hh + h];
    float ln = (v - sMu[bp]) * sRs[bp] * w + bb;
#pragma unroll
    for (int q = 0; q < Pp; ++q) out[q] += sW[q * Pp + p] * ln;
  }
  const float aw = attn_w(attn, blk + 1);
#pragma unroll
  for (int q = 0; q < Pp; ++q) {
    size_t o = (((size_t)(b * Pp + q)) * Nn + n) * Hh + h;
    Xn[o]   = out[q];
    ACC[o] += aw * out[q];
  }
}

// ---------------------------------------------------------------------------
// output projection: out = relu(ACC@ow1^T+ob1)@ow2^T + ob2   (WMMA, TDM)
// ---------------------------------------------------------------------------
__global__ __launch_bounds__(256)
void k_proj_out(const float* __restrict__ ACC,
                const _Float16* __restrict__ w1, const float* __restrict__ b1,
                const _Float16* __restrict__ w2, const float* __restrict__ b2,
                float* __restrict__ out) {
  __shared__ _Float16 sW1[64 * 64];
  __shared__ _Float16 sW2[64 * 64];
  __shared__ _Float16 sA[128 * 72];
  __shared__ _Float16 sH[128 * 72];
  const int tid = threadIdx.x;
  if (tid == 0) { tdm_load_w64(w1, sW1); tdm_load_w64(w2, sW2); }
  const int row0 = blockIdx.x * 128;
  for (int idx = tid; idx < 128 * 64; idx += 256) {
    int r = idx >> 6, cc = idx & 63;
    sA[r * 72 + cc] = (_Float16)ACC[(size_t)(row0 + r) * 64 + cc];
  }
  __builtin_amdgcn_s_wait_tensorcnt(0);
  __syncthreads();
  const int wave = tid >> 5, lane = tid & 31;
  const int arow0 = wave * 16, half = lane >> 4, nlo = lane & 15;
  v8f c[4] = {};
  wave_gemm_16x64(sA, 72, arow0, sW1, c);
#pragma unroll
  for (int nt = 0; nt < 4; ++nt) {
#pragma unroll
    for (int r = 0; r < 8; ++r) {
      int rl = arow0 + half * 8 + r, col = nt * 16 + nlo;
      sH[rl * 72 + col] = (_Float16)fmaxf(c[nt][r] + b1[col], 0.f);
    }
  }
  __syncthreads();
  v8f d[4] = {};
  wave_gemm_16x64(sH, 72, arow0, sW2, d);
#pragma unroll
  for (int nt = 0; nt < 4; ++nt) {
#pragma unroll
    for (int r = 0; r < 8; ++r) {
      int rl = arow0 + half * 8 + r, col = nt * 16 + nlo;
      out[(size_t)(row0 + rl) * 64 + col] = d[nt][r] + b2[col];
    }
  }
}

// ---------------------------------------------------------------------------
extern "C" void kernel_launch(void* const* d_in, const int* in_sizes, int n_in,
                              void* d_out, int out_size, void* d_ws, size_t ws_size,
                              hipStream_t stream) {
  const float*     x    = (const float*)d_in[0];
  const long long* ei   = (const long long*)d_in[1];
  const float*     pos  = (const float*)d_in[2];
  const float*     iw1  = (const float*)d_in[3];
  const float*     ib1  = (const float*)d_in[4];
  const float*     iw2  = (const float*)d_in[5];
  const float*     ib2  = (const float*)d_in[6];
  const float*     pw1  = (const float*)d_in[7];
  const float*     pb1  = (const float*)d_in[8];
  const float*     pw2  = (const float*)d_in[9];
  const float*     pb2  = (const float*)d_in[10];
  const float*     attn = (const float*)d_in[11];
  const float*     gnw  = (const float*)d_in[12];
  const float*     gnb  = (const float*)d_in[13];
  const float*     tnw  = (const float*)d_in[14];
  const float*     tnb  = (const float*)d_in[15];
  const float*     tcw  = (const float*)d_in[16];
  const float*     tcb  = (const float*)d_in[17];
  const float*     ow1  = (const float*)d_in[18];
  const float*     ob1  = (const float*)d_in[19];
  const float*     ow2  = (const float*)d_in[20];
  const float*     ob2  = (const float*)d_in[21];
  float* out = (float*)d_out;

  // Workspace carve-out (256B aligned regions)
  char* base = (char*)d_ws;
  size_t off = 0;
  auto carve = [&](size_t bytes) -> void* {
    void* p = base + off;
    off = (off + bytes + 255) & ~(size_t)255;
    return p;
  };
  float*     deg  = (float*)carve((size_t)Nn * 4);
  float*     nrm  = (float*)carve((size_t)Ee * 4);
  int*       rptr = (int*)carve((size_t)(Nn + 1) * 4);
  int*       cur  = (int*)carve((size_t)Nn * 4);
  int*       csrc = (int*)carve((size_t)Ee * 4);
  float*     cw   = (float*)carve((size_t)Ee * 4);
  float*     posP = (float*)carve((size_t)Nn * Hh * 4);
  _Float16*  wf16 = (_Float16*)carve((size_t)5 * 4096 * 2);
  float*     part = (float*)carve((size_t)BPc * SBLK * 2 * 4);
  float*     st1  = (float*)carve((size_t)BPc * 2 * 4);
  float*     st2  = (float*)carve((size_t)BPc * 2 * 4);
  float*     X    = (float*)carve((size_t)ROWS * Hh * 4);
  float*     XB   = (float*)carve((size_t)ROWS * Hh * 4);
  float*     ACC  = (float*)carve((size_t)ROWS * Hh * 4);
  (void)ws_size; (void)in_sizes; (void)n_in; (void)out_size;

  // Graph prep + weight conversion
  k_zero<<<(Nn + 255) / 256, 256, 0, stream>>>(deg, Nn);
  k_cvt_w<<<(5 * 4096 + 255) / 256, 256, 0, stream>>>(iw2, pw1, pw2, ow1, ow2, wf16);
  k_deg<<<(Ee + 255) / 256, 256, 0, stream>>>(ei, deg);
  k_norm<<<(Ee + 255) / 256, 256, 0, stream>>>(ei, deg, nrm);
  k_scan<<<1, 1024, 0, stream>>>(deg, rptr);
  k_csr_init<<<(Nn + 255) / 256, 256, 0, stream>>>(rptr, cur);
  k_fill<<<(Ee + 255) / 256, 256, 0, stream>>>(ei, nrm, cur, csrc, cw);

  // Projections (WMMA + TDM)
  k_pos_proj<<<(Nn + 127) / 128, 256, 0, stream>>>(pos, wf16 + 1 * 4096, pb1,
                                                   wf16 + 2 * 4096, pb2, posP);
  k_proj_in<<<ROWS / 128, 256, 0, stream>>>(x, iw1, ib1, wf16 + 0 * 4096, ib2,
                                            posP, attn, X, ACC);

  // 3 LightGCN blocks
  for (int i = 0; i < NBl; ++i) {
    k_stats<<<dim3(SBLK, BPc), 256, 0, stream>>>(X, part);
    k_stats_final<<<BPc, 256, 0, stream>>>(part, st1);
    k_propagate<<<dim3(Nn, BPc / 4), 256, 0, stream>>>(
        X, rptr, csrc, cw, st1, gnw + (size_t)i * SLICE, gnb + (size_t)i * SLICE, XB);
    k_stats<<<dim3(SBLK, BPc), 256, 0, stream>>>(XB, part);
    k_stats_final<<<BPc, 256, 0, stream>>>(part, st2);
    k_tconv<<<(Bb * Nn * Hh + 255) / 256, 256, 0, stream>>>(
        XB, st2, tnw + (size_t)i * SLICE, tnb + (size_t)i * SLICE,
        tcw + i * Pp * Pp, tcb + i * Pp, attn, i, X, ACC);
  }

  // Output projection
  k_proj_out<<<ROWS / 128, 256, 0, stream>>>(ACC, wf16 + 3 * 4096, ob1,
                                             wf16 + 4 * 4096, ob2, out);
}
